// ResRnn_66451734004183
// MI455X (gfx1250) — compile-verified, matrix operand order checked by reading, not run on hardware
//
#include <hip/hip_runtime.h>
#include <math.h>

#define SEQ   2048
#define BATCH 256
#define INW   64
#define SW    512
#define OUTW  256
#define IP    0.97f

typedef _Float16 f16;
typedef __attribute__((ext_vector_type(16))) _Float16 v16h;
typedef __attribute__((ext_vector_type(8)))  float    v8f;
typedef __attribute__((ext_vector_type(4)))  unsigned int v4u;

// global (address_space(1)) flavor of the 16-byte chunk type, so weight
// loads lower to global_load_b128 (LOADcnt only) instead of flat_load_b128.
typedef v4u __attribute__((address_space(1))) gv4u;

union HPack { v4u q[2]; v16h v; };

// padded LDS row strides (elements) to avoid 16-way bank conflicts on the
// strided A-operand loads (row stride must not be a multiple of 64 dwords)
#define RS16 520   // f16 elements  -> 1040 B rows (16B aligned, 260 dwords)
#define RS32 516   // f32 elements  -> 2064 B rows

// ---------------------------------------------------------------------------
// Prep: convert W1/W2 (fp32, [out=512][in=512]) to fp16 in workspace.
// ---------------------------------------------------------------------------
__global__ void wcvt_kernel(const float* __restrict__ W1,
                            const float* __restrict__ W2,
                            f16* __restrict__ W1h,
                            f16* __restrict__ W2h) {
    int i = blockIdx.x * blockDim.x + threadIdx.x;
    if (i < SW * SW) {
        W1h[i] = (f16)W1[i];
        W2h[i] = (f16)W2[i];
    }
}

// ---------------------------------------------------------------------------
// Persistent per-row-block recurrence. grid = 16 WGs (one 16-row batch tile
// each), block = 1024 threads = 32 waves; wave w owns output N-tile w.
// No cross-WG dependency exists: the K reductions run over features only.
// ---------------------------------------------------------------------------
__global__ __launch_bounds__(1024, 1)
void resrnn_kernel(const float* __restrict__ x,    // [SEQ][BATCH][INW]
                   const f16*   __restrict__ W1h,  // [SW][SW]
                   const float* __restrict__ b1,   // [SW]
                   const f16*   __restrict__ W2h,  // [SW][SW]
                   const float* __restrict__ b2,   // [SW]
                   float*       __restrict__ out)  // [BATCH][OUTW]
{
    __shared__ __align__(16) f16   abH[16 * RS16]; // r (f16) then h (f16)
    __shared__ __align__(16) float rF [16 * RS32]; // fp32 master state r

    const int tid  = threadIdx.x;
    const int lane = tid & 31;
    const int wid  = tid >> 5;              // 0..31 == N-tile index
    const int row0 = blockIdx.x * 16;       // batch row base of this WG

    const int m    = lane & 15;             // A-operand row within tile
    const int hi   = lane >> 4;             // lane-half selects K sub-chunk
    const int ncol = (wid << 4) + (lane & 15); // this lane's output column n
    const int abase = m * RS16 + hi * 8;    // A-operand LDS base (elements)

    const float b1v = b1[ncol];
    const float b2v = b2[ncol];

    // r0 = [x_0, zeros]
    for (int idx = tid; idx < 16 * SW; idx += 1024) {
        int mm = idx >> 9;
        int c  = idx & 511;
        float v = (c < INW) ? x[(size_t)(row0 + mm) * INW + c] : 0.0f;
        rF [mm * RS32 + c] = v;
        abH[mm * RS16 + c] = (f16)v;
    }
    __syncthreads();

    // weight row base addresses as opaque u64 (defeats LICM of 1 MB of
    // B-operand loads out of the t-loop, which would cause scratch spills).
    unsigned long long w1a = (unsigned long long)(W1h + (size_t)ncol * SW);
    unsigned long long w2a = (unsigned long long)(W2h + (size_t)ncol * SW);

    // 2-deep software-pipelined 16x16x512 GEMM slice. Both 16-B B-chunks are
    // expressed directly off one per-lane base address so they lower to
    // immediate-offset global_load_b128 (no per-chunk v_add_nc_u64 / xcnt).
#define GEMM512(ACC, WBASE)                                                  \
    {                                                                        \
        const unsigned long long wb = (WBASE)                                \
                                    + (unsigned long long)(hi * 16);         \
        HPack a0, a1, a2, b0, b1x, b2x;                                      \
        a0.q[0] = *(const v4u*)&abH[abase];                                  \
        a0.q[1] = *(const v4u*)&abH[abase + 16];                             \
        b0.q[0] = *(const gv4u*)(wb);                                        \
        b0.q[1] = *(const gv4u*)(wb + 32);                                   \
        a1.q[0] = *(const v4u*)&abH[abase + 32];                             \
        a1.q[1] = *(const v4u*)&abH[abase + 48];                             \
        b1x.q[0] = *(const gv4u*)(wb + 64);                                  \
        b1x.q[1] = *(const gv4u*)(wb + 96);                                  \
        a2 = a0; b2x = b0;                                                   \
        _Pragma("unroll")                                                    \
        for (int ks = 0; ks < 16; ++ks) {                                    \
            if (ks + 2 < 16) {                                               \
                const int kn = (ks + 2) * 32;                                \
                a2.q[0] = *(const v4u*)&abH[abase + kn];                     \
                a2.q[1] = *(const v4u*)&abH[abase + kn + 16];                \
                b2x.q[0] = *(const gv4u*)(wb + (unsigned long long)(kn * 2));\
                b2x.q[1] = *(const gv4u*)(wb +                               \
                               (unsigned long long)(kn * 2 + 32));           \
            }                                                                \
            ACC = __builtin_amdgcn_wmma_f32_16x16x32_f16(                    \
                      false, a0.v, false, b0.v, (short)0, ACC,               \
                      false, false);                                         \
            a0 = a1; a1 = a2; b0 = b1x; b1x = b2x;                           \
        }                                                                    \
    }

    for (int t = 0; t < SEQ; ++t) {
        asm volatile("" : "+v"(w1a), "+v"(w2a));

        // Hoist end-of-step latency: residual sources and next x are stable
        // until this step's write stage, so read them now and let the loads
        // overlap GEMM1.
        float rsrc[8];
        #pragma unroll
        for (int v = 0; v < 8; ++v)
            rsrc[v] = rF[(v + hi * 8) * RS32 + ncol];
        float xv = 0.0f;
        if (t < SEQ - 1)
            xv = x[((size_t)(t + 1) * BATCH + (row0 + (tid >> 6))) * INW
                   + (tid & 63)];

        // ---- GEMM1: acc = r @ W1^T  (A from LDS, B streamed from L2) ----
        v8f acc = {};
        GEMM512(acc, w1a);
        __syncthreads();                    // everyone done reading r (f16)

        // h = |acc + b1| -> overwrite abH with h tile
        #pragma unroll
        for (int v = 0; v < 8; ++v) {
            int mr = v + hi * 8;            // C/D layout: M = v + 8*laneHi
            abH[mr * RS16 + ncol] = (f16)fabsf(acc[v] + b1v);
        }
        __syncthreads();                    // h fully materialized

        // ---- GEMM2: acc2 = h @ W2^T ----
        v8f acc2 = {};
        GEMM512(acc2, w2a);

        // residual in fp32 against the prefetched fp32 master state
        float nv[8];
        #pragma unroll
        for (int v = 0; v < 8; ++v)
            nv[v] = IP * rsrc[v] + (1.0f - IP) * (acc2[v] + b2v);
        __syncthreads();    // all rF/abH(h) reads done before rewrite

        if (t == SEQ - 1) {
            if (ncol >= OUTW) {             // wave-uniform (wid >= 16)
                #pragma unroll
                for (int v = 0; v < 8; ++v) {
                    int mr = v + hi * 8;
                    out[(size_t)(row0 + mr) * OUTW + (ncol - OUTW)] = nv[v];
                }
            }
        } else {
            // shift: r_next[:, n+64] = new_state[:, n]  (n < 448)
            if (ncol < SW - INW) {          // wave-uniform (wid < 28)
                #pragma unroll
                for (int v = 0; v < 8; ++v) {
                    int mr = v + hi * 8;
                    rF [mr * RS32 + ncol + INW] = nv[v];
                    abH[mr * RS16 + ncol + INW] = (f16)nv[v];
                }
            }
            // r_next[:, 0:64] = x_{t+1}  (prefetched into xv)
            {
                int mm = tid >> 6;
                int c  = tid & 63;
                rF [mm * RS32 + c] = xv;
                abH[mm * RS16 + c] = (f16)xv;
            }
        }
        __syncthreads();                    // r_next ready for next GEMM1
    }
#undef GEMM512
}

// ---------------------------------------------------------------------------
extern "C" void kernel_launch(void* const* d_in, const int* in_sizes, int n_in,
                              void* d_out, int out_size, void* d_ws, size_t ws_size,
                              hipStream_t stream) {
    (void)in_sizes; (void)n_in; (void)out_size; (void)ws_size;
    const float* x  = (const float*)d_in[0];
    const float* W1 = (const float*)d_in[1];
    const float* b1 = (const float*)d_in[2];
    const float* W2 = (const float*)d_in[3];
    const float* b2 = (const float*)d_in[4];
    float* out = (float*)d_out;

    f16* W1h = (f16*)d_ws;                  // 512 KB
    f16* W2h = W1h + (size_t)SW * SW;       // 512 KB

    wcvt_kernel<<<(SW * SW + 255) / 256, 256, 0, stream>>>(W1, W2, W1h, W2h);
    resrnn_kernel<<<BATCH / 16, 1024, 0, stream>>>(x, W1h, b1, W2h, b2, out);
}
